// EntitiesAssignment_85968065397426
// MI455X (gfx1250) — matrix operand order repaired and checked by reading, not compiled, not run on hardware
//
#include <hip/hip_runtime.h>

// CDNA5 (gfx1250) wave32 WMMA types
typedef __attribute__((ext_vector_type(16))) _Float16 v16h;
typedef __attribute__((ext_vector_type(8)))  float    v8f;

#define WAVES_PER_BLOCK 8   // 256 threads = 8 wave32; 2 batches per wave -> 16 batches/block

// ---------------------------------------------------------------------------
// Kernel 1: per-wave, compute two 8x8 cost matrices via one 16x16 WMMA tile
// (split-f16, 6 x v_wmma_f32_16x16x32_f16 for fp32-equivalent precision),
// then Held-Karp assignment DP (256 masks, lane-parallel min-plus), then a
// fixed-order per-block partial sum.
// ---------------------------------------------------------------------------
__global__ __launch_bounds__(256)
void pit_wmma_dp_kernel(const float* __restrict__ yt,   // [B,8,64]
                        const float* __restrict__ yp,   // [B,8,64]
                        float* __restrict__ partial,    // [gridDim.x]
                        int B)
{
    __shared__ float dp[WAVES_PER_BLOCK][2][256];   // Held-Karp dp per (wave, batch-in-wave)
    __shared__ float wsum[WAVES_PER_BLOCK * 2];     // per-batch best/8

    const int tid  = threadIdx.x;
    const int lane = tid & 31;
    const int wave = tid >> 5;

    if (tid < WAVES_PER_BLOCK * 2) wsum[tid] = 0.0f;
    __syncthreads();

    const int waveGlobal = blockIdx.x * WAVES_PER_BLOCK + wave;
    const int b0 = 2 * waveGlobal;
    const bool wave_valid = (b0 < B);
    const bool g1ok = (b0 + 1) < B;
    const int b1 = g1ok ? (b0 + 1) : b0;

    if (wave_valid) {
        // Tile row (A) / tile col (B) handled by this lane; rows 0-7 -> batch b0,
        // rows 8-15 -> batch b1.
        const int mrow = lane & 15;
        const int hi   = lane >> 4;                // half-wave select
        const int bat  = (mrow < 8) ? b0 : b1;
        const float* trow = yt + ((size_t)bat * 8 + (mrow & 7)) * 64;
        const float* prow = yp + ((size_t)bat * 8 + (mrow & 7)) * 64;

        // ISA 16-bit A layout: lanes 0-15 own K = {0..7, 16..23} (+32 for 2nd wmma),
        // lanes 16-31 own K = {8..15, 24..31}. B layout: same as A,
        // +32 for 2nd wmma.
        const int ka = hi ? 8 : 0;
        const int kb = hi ? 8 : 0;

        float ta[32], pb[32];
        #pragma unroll
        for (int c = 0; c < 4; ++c) {              // A chunks at ka + 16*c, 8 floats each
            float4 u = *(const float4*)(trow + ka + 16 * c);
            float4 v = *(const float4*)(trow + ka + 16 * c + 4);
            ta[8*c+0]=u.x; ta[8*c+1]=u.y; ta[8*c+2]=u.z; ta[8*c+3]=u.w;
            ta[8*c+4]=v.x; ta[8*c+5]=v.y; ta[8*c+6]=v.z; ta[8*c+7]=v.w;
        }
        #pragma unroll
        for (int c = 0; c < 2; ++c)                // B chunks at kb + 32*c, 16 floats each
            #pragma unroll
            for (int q = 0; q < 4; ++q) {
                float4 v = *(const float4*)(prow + kb + 32 * c + 4 * q + 8 * (q >> 1));
                pb[16*c+4*q+0]=v.x; pb[16*c+4*q+1]=v.y;
                pb[16*c+4*q+2]=v.z; pb[16*c+4*q+3]=v.w;
            }

        // Exact fp32 row norms: lane and lane^16 load complementary halves.
        float tn = 0.f, pn = 0.f;
        #pragma unroll
        for (int i = 0; i < 32; ++i) { tn += ta[i] * ta[i]; pn += pb[i] * pb[i]; }
        tn += __shfl_xor(tn, 16, 32);
        pn += __shfl_xor(pn, 16, 32);

        // Split-f16: x = hi + lo, keep hi*hi + hi*lo + lo*hi (error ~2^-22 rel).
        v16h ah0, ah1, al0, al1, bh0, bh1, bl0, bl1;
        #pragma unroll
        for (int i = 0; i < 16; ++i) {
            float x0 = ta[i], x1 = ta[16 + i];
            _Float16 h0 = (_Float16)x0, h1 = (_Float16)x1;
            ah0[i] = h0; al0[i] = (_Float16)(x0 - (float)h0);
            ah1[i] = h1; al1[i] = (_Float16)(x1 - (float)h1);
            float y0 = pb[i], y1 = pb[16 + i];
            _Float16 g0 = (_Float16)y0, g1 = (_Float16)y1;
            bh0[i] = g0; bl0[i] = (_Float16)(y0 - (float)g0);
            bh1[i] = g1; bl1[i] = (_Float16)(y1 - (float)g1);
        }

        // G = T * P^T for both batches in one 16x16 tile, K=64 in two K=32 steps.
        v8f acc = {};
        acc = __builtin_amdgcn_wmma_f32_16x16x32_f16(false, ah0, false, bh0, (short)0, acc, false, false);
        acc = __builtin_amdgcn_wmma_f32_16x16x32_f16(false, ah1, false, bh1, (short)0, acc, false, false);
        acc = __builtin_amdgcn_wmma_f32_16x16x32_f16(false, al0, false, bh0, (short)0, acc, false, false);
        acc = __builtin_amdgcn_wmma_f32_16x16x32_f16(false, al1, false, bh1, (short)0, acc, false, false);
        acc = __builtin_amdgcn_wmma_f32_16x16x32_f16(false, ah0, false, bl0, (short)0, acc, false, false);
        acc = __builtin_amdgcn_wmma_f32_16x16x32_f16(false, ah1, false, bl1, (short)0, acc, false, false);

        // C/D layout: VGPR r holds row M=r (lanes 0-15) or M=8+r (lanes 16-31),
        // col N = lane&15. Active columns: lanes 0-7 (b0), lanes 24-31 (b1).
        // cost[i] = (|t_i|^2 + |p_j|^2 - 2*G[i][j]) / 64
        float cost[8];
        #pragma unroll
        for (int i = 0; i < 8; ++i) {
            float tni = __shfl(tn, (lane & 24) + i, 32);
            cost[i] = (tni + pn - 2.0f * acc[i]) * (1.0f / 64.0f);
        }

        const bool active = (lane < 8) || (lane >= 24);
        const int  grp = (lane >= 16) ? 1 : 0;
        const int  j   = lane & 7;
        float* dpw = &dp[wave][grp][0];

        if (active) {
            float best = 0.0f;
            if (j == 0) dpw[0] = 0.0f;
            // Held-Karp: level k assigns rows 0..k-1; masks enumerated by Gosper's hack.
            #pragma unroll
            for (int k = 1; k <= 8; ++k) {
                unsigned m = (1u << k) - 1u;
                while (m < 256u) {
                    float dv = dpw[m & ~(1u << j)];          // garbage if bit unset; discarded
                    float cand = ((m >> j) & 1u) ? (dv + cost[k - 1]) : 3.0e38f;
                    cand = fminf(cand, __shfl_xor(cand, 1, 32));
                    cand = fminf(cand, __shfl_xor(cand, 2, 32));
                    cand = fminf(cand, __shfl_xor(cand, 4, 32));
                    if (j == 0) dpw[m] = cand;               // DS in-order within wave
                    if (m == 255u) best = cand;
                    unsigned lo = m & (unsigned)(-(int)m);   // Gosper next-combination
                    unsigned r  = m + lo;
                    m = r | (((m ^ r) >> 2) / lo);
                }
            }
            if (j == 0) {
                float v = best * 0.125f;                     // divide by N=8
                if (grp == 1 && !g1ok) v = 0.0f;
                wsum[wave * 2 + grp] = v;
            }
        }
    }

    __syncthreads();
    if (tid == 0) {                                          // fixed-order, deterministic
        float s = 0.0f;
        #pragma unroll
        for (int i = 0; i < WAVES_PER_BLOCK * 2; ++i) s += wsum[i];
        partial[blockIdx.x] = s;
    }
}

// ---------------------------------------------------------------------------
// Kernel 2: deterministic fixed-tree reduction of per-block partials -> mean.
// ---------------------------------------------------------------------------
__global__ __launch_bounds__(128)
void pit_reduce_kernel(const float* __restrict__ partial, float* __restrict__ out,
                       int nb, float scale)
{
    __shared__ float s[128];
    const int t = threadIdx.x;
    float a = 0.0f;
    for (int i = t; i < nb; i += 128) a += partial[i];
    s[t] = a;
    __syncthreads();
    #pragma unroll
    for (int off = 64; off > 0; off >>= 1) {
        if (t < off) s[t] += s[t + off];
        __syncthreads();
    }
    if (t == 0) out[0] = s[0] * scale;
}

extern "C" void kernel_launch(void* const* d_in, const int* in_sizes, int n_in,
                              void* d_out, int out_size, void* d_ws, size_t ws_size,
                              hipStream_t stream)
{
    const float* yt = (const float*)d_in[0];   // y_true [B,8,64] fp32
    const float* yp = (const float*)d_in[1];   // y_pred [B,8,64] fp32
    float* out = (float*)d_out;                // scalar fp32

    const int B = in_sizes[0] / (8 * 64);      // 2048
    const int batchesPerBlock = 2 * WAVES_PER_BLOCK;
    const int nb = (B + batchesPerBlock - 1) / batchesPerBlock;

    float* partial = (float*)d_ws;             // nb floats of scratch

    pit_wmma_dp_kernel<<<nb, 256, 0, stream>>>(yt, yp, partial, B);
    pit_reduce_kernel<<<1, 128, 0, stream>>>(partial, out, nb, 1.0f / (float)B);
}
